// ChannelMixingKAN_16303695855780
// MI455X (gfx1250) — compile-verified, matrix operand order checked by reading, not run on hardware
//
#include <hip/hip_runtime.h>
#include <hip/hip_bf16.h>

// ---------------------------------------------------------------------------
// ChannelMixingKAN for MI455X (gfx1250), wave32 + WMMA f16->f32.
//
// KAN layers as f16 GEMMs over expanded features:
//   L1: A1 (N x 1152) @ W1eff^T (1152 x 256)  features = [silu(x) | spline8(x)]
//   L2: A2 (N x 2304) @ W2eff^T (2304 x 128)  features = [silu(h) | spline8(h)]
// M=32 rows per block, 2x2 (L1) / 2x1 (L2) register blocking so each
// pre-swizzled B fragment (one 32B global load / lane) feeds 2 WMMAs.
// Activation tiles live in LDS with bank-conflict-free padded strides; A1 is
// aliased inside the A2 region. Spline bases use the closed-form cardinal
// cubic (only 4 non-zero bases) instead of full Cox-de-Boor.
// ---------------------------------------------------------------------------

typedef __attribute__((ext_vector_type(16))) _Float16 v16h;
typedef __attribute__((ext_vector_type(8)))  _Float16 h8;
typedef __attribute__((ext_vector_type(8)))  float    v8f;

#define BB   64
#define CC   128
#define TT   512
#define HH   256
#define K1   1152            // 128 + 128*8
#define K2   2304            // 256 + 256*8
#define KO1  36              // K1/32
#define KO2  72              // K2/32
#define MT   32              // rows per block (512 % 32 == 0)

// Padded LDS strides (elements). Chosen so row step mod 64 banks is 4 / 4 / 1:
#define A1S  1160            // halfs  (2320 B, /4 = 580 dw, 580%64 = 4)
#define A2S  2312            // halfs  (4624 B, /4 = 1156 dw, 1156%64 = 4)
#define HBS  257             // floats (257%64 = 1)

#define LDS_A2_BYTES   (MT * A2S * 2)                 // 147968 (A1 aliased here)
#define LDS_HB_OFF     LDS_A2_BYTES
#define LDS_TOTAL      (LDS_A2_BYTES + MT * HBS * 4)  // 180864

__device__ __forceinline__ float silu_f(float x) {
    return x * (1.0f / (1.0f + __expf(-x)));
}

__device__ __forceinline__ v8f wmma16(v16h a, v16h b, v8f c) {
    return __builtin_amdgcn_wmma_f32_16x16x32_f16(
        false, a, false, b, (short)0, c, false, false);
}

// 16-bit A-fragment from LDS: lane holds row = lane%16; halfs 0..7 at
// K = 8*(lane/16), halfs 8..15 at K = 16 + 8*(lane/16) in each 32-wide block.
__device__ __forceinline__ v16h load_frag16(const _Float16* p) {
    union { v16h v; h8 h[2]; } u;
    u.h[0] = *(const h8*)(p);
    u.h[1] = *(const h8*)(p + 16);
    return u.v;
}

// Cubic B-spline bases on the uniform KAN grid (h=0.4, knots (j-3)*0.4-1).
// Only bases j in [i-3, i] are non-zero for x in knot interval i; write the
// 8-slot f16 group as one zeroing b128 store + <=4 predicated b16 stores.
__device__ __forceinline__ void bspline8_store(float x, _Float16* dst8) {
    const float u  = 2.5f * x + 5.5f;          // (x+1)/0.4 + 3
    const float fi = floorf(u);
    const float w  = u - fi;
    const int   ii = (int)fi;
    const float om = 1.0f - w;
    const float w2 = w * w;
    const float w3 = w2 * w;
    const float p0 = om * om * om * (1.0f / 6.0f);
    const float p1 = 0.5f * w3 - w2 + (2.0f / 3.0f);
    const float p2 = -0.5f * w3 + 0.5f * w2 + 0.5f * w + (1.0f / 6.0f);
    const float p3 = w3 * (1.0f / 6.0f);
    h8 z = {};
    *(h8*)dst8 = z;                            // zero all 8 slots
    const bool valid = (ii >= 0) && (ii <= 10);
    const int  s = ii - 3;
    if (valid) {
        int j;
        j = s + 0; if (j >= 0 && j < 8) dst8[j] = (_Float16)p0;
        j = s + 1; if (j >= 0 && j < 8) dst8[j] = (_Float16)p1;
        j = s + 2; if (j >= 0 && j < 8) dst8[j] = (_Float16)p2;
        j = s + 3; if (j >= 0 && j < 8) dst8[j] = (_Float16)p3;
    }
}

// -------------------------- per-batch LayerNorm stats ----------------------
__global__ void __launch_bounds__(256)
stats_kernel(const float* __restrict__ U, float* __restrict__ mu,
             float* __restrict__ rsig) {
    __shared__ float ssum[256];
    __shared__ float ssq[256];
    const int b = blockIdx.x, tid = threadIdx.x;
    const float* p = U + (size_t)b * (CC * TT);
    float s = 0.f, q = 0.f;
#pragma unroll 4
    for (int i = 0; i < (CC * TT) / 256; ++i) {
        float v = p[tid + 256 * i];
        s += v; q += v * v;
    }
    ssum[tid] = s; ssq[tid] = q;
    __syncthreads();
    for (int off = 128; off > 0; off >>= 1) {
        if (tid < off) { ssum[tid] += ssum[tid + off]; ssq[tid] += ssq[tid + off]; }
        __syncthreads();
    }
    if (tid == 0) {
        float m = ssum[0] * (1.0f / (CC * TT));
        float v = ssq[0] * (1.0f / (CC * TT)) - m * m;
        mu[b] = m;
        rsig[b] = rsqrtf(v + 1e-5f);
    }
}

// K index inside a 32-wide block for B-fragment half t on lane-half hi.
__device__ __forceinline__ int frag_k(int hi, int t) {
    return (t < 8) ? (hi * 8 + t) : (16 + hi * 8 + (t - 8));
}

// ------------------ fuse + swizzle weights into B-fragment order -----------
// W1f layout: [ko (36)][ntile (16)][lane (32)][half (16)]  (f16)
__global__ void __launch_bounds__(256)
prep_w1(const float* __restrict__ bw1, const float* __restrict__ sw1,
        const float* __restrict__ ss1, _Float16* __restrict__ W1f) {
    int idx = blockIdx.x * blockDim.x + threadIdx.x;       // over 36*16*32
    if (idx >= KO1 * 16 * 32) return;
    int lane = idx & 31;
    int nt   = (idx >> 5) & 15;
    int ko   = idx >> 9;
    int n    = nt * 16 + (lane & 15);                       // output row (H)
    int hi   = lane >> 4;
    _Float16* dst = W1f + (size_t)idx * 16;
#pragma unroll
    for (int t = 0; t < 16; ++t) {
        int k = ko * 32 + frag_k(hi, t);
        float w;
        if (k < CC) {
            w = bw1[n * CC + k];
        } else {
            int q = k - CC, c = q >> 3, j = q & 7;
            w = sw1[(n * CC + c) * 8 + j] * ss1[n * CC + c];
        }
        dst[t] = (_Float16)w;
    }
}

// W2f layout: [ko (72)][ntile (8)][lane (32)][half (16)]  (f16)
__global__ void __launch_bounds__(256)
prep_w2(const float* __restrict__ bw2, const float* __restrict__ sw2,
        const float* __restrict__ ss2, _Float16* __restrict__ W2f) {
    int idx = blockIdx.x * blockDim.x + threadIdx.x;       // over 72*8*32
    if (idx >= KO2 * 8 * 32) return;
    int lane = idx & 31;
    int nt   = (idx >> 5) & 7;
    int ko   = idx >> 8;
    int n    = nt * 16 + (lane & 15);                       // output row (C)
    int hi   = lane >> 4;
    _Float16* dst = W2f + (size_t)idx * 16;
#pragma unroll
    for (int t = 0; t < 16; ++t) {
        int k = ko * 32 + frag_k(hi, t);
        float w;
        if (k < HH) {
            w = bw2[n * HH + k];
        } else {
            int q = k - HH, h = q >> 3, j = q & 7;
            w = sw2[(n * HH + h) * 8 + j] * ss2[n * HH + h];
        }
        dst[t] = (_Float16)w;
    }
}

// ------------------------------ fused main kernel --------------------------
// One block = 32 rows of one batch (512 % 32 == 0, b uniform). 8 waves.
__global__ void __launch_bounds__(256)
kan_main(const float* __restrict__ U, const float* __restrict__ ln_w,
         const float* __restrict__ ln_b, const float* __restrict__ mu,
         const float* __restrict__ rsig, const v16h* __restrict__ W1f,
         const v16h* __restrict__ W2f, float* __restrict__ out) {
    extern __shared__ char smem[];
    _Float16* A1 = (_Float16*)smem;                 // 32 x K1 (stride A1S), dies after L1
    _Float16* A2 = (_Float16*)smem;                 // 32 x K2 (stride A2S), aliases A1
    float*    Hb = (float*)(smem + LDS_HB_OFF);     // 32 x 256 (stride HBS)

    const int tid  = threadIdx.x;
    const int lane = tid & 31;
    const int wave = tid >> 5;
    const int nlow = lane & 15;
    const int hi   = lane >> 4;

    const int r0 = blockIdx.x * MT;
    const int b  = r0 >> 9;          // r0 / 512
    const int t0 = r0 & 511;

    const float m_mu = mu[b];
    const float m_rs = rsig[b];

    // ---- Phase A: LN + silu + spline bases -> A1 (32 x 1152 f16) ----
    {
        const int m  = tid & 31;     // row -> contiguous t across a wave
        const int cg = tid >> 5;     // 0..7
        const int t  = t0 + m;
        _Float16* arow = A1 + (size_t)m * A1S;
#pragma unroll
        for (int i = 0; i < 16; ++i) {
            const int c = cg + 8 * i;
            float x  = U[((size_t)b * CC + c) * TT + t];
            float xn = (x - m_mu) * m_rs * ln_w[t * CC + c] + ln_b[t * CC + c];
            arow[c] = (_Float16)silu_f(xn);
            bspline8_store(xn, arow + CC + c * 8);
        }
    }
    __syncthreads();

    // ---- Layer-1 GEMM: (32 x 1152) x (1152 x 256); wave -> 2 M x 2 N ----
    {
        const int nt0 = wave * 2, nt1 = nt0 + 1;
        v8f c00 = {}, c01 = {}, c10 = {}, c11 = {};  // [mt][nt]
        const _Float16* ar0 = A1 + (size_t)nlow * A1S + hi * 8;
        const _Float16* ar1 = A1 + (size_t)(16 + nlow) * A1S + hi * 8;
#pragma unroll 2
        for (int ko = 0; ko < KO1; ++ko) {
            v16h a0 = load_frag16(ar0 + ko * 32);
            v16h a1 = load_frag16(ar1 + ko * 32);
            v16h b0 = W1f[(ko * 16 + nt0) * 32 + lane];
            v16h b1 = W1f[(ko * 16 + nt1) * 32 + lane];
            c00 = wmma16(a0, b0, c00);
            c10 = wmma16(a1, b0, c10);
            c01 = wmma16(a0, b1, c01);
            c11 = wmma16(a1, b1, c11);
        }
        // C/D layout: element v of lane -> (M = v + 8*hi, N = nlow)
#pragma unroll
        for (int v = 0; v < 8; ++v) {
            const int rA = (v + 8 * hi) * HBS;
            const int rB = (16 + v + 8 * hi) * HBS;
            Hb[rA + nt0 * 16 + nlow] = c00[v];
            Hb[rA + nt1 * 16 + nlow] = c01[v];
            Hb[rB + nt0 * 16 + nlow] = c10[v];
            Hb[rB + nt1 * 16 + nlow] = c11[v];
        }
    }
    __syncthreads();   // A1 fully consumed; its LDS now becomes A2

    // ---- Phase B: silu + spline bases of h -> A2 (32 x 2304 f16) ----
    {
        const int m  = tid & 31;
        const int hg = tid >> 5;     // 0..7
        const float* hrow = Hb + (size_t)m * HBS;
        _Float16* arow = A2 + (size_t)m * A2S;
#pragma unroll
        for (int i = 0; i < 32; ++i) {
            const int hc = hg + 8 * i;
            float h = hrow[hc];
            arow[hc] = (_Float16)silu_f(h);
            bspline8_store(h, arow + HH + hc * 8);
        }
    }
    __syncthreads();

    // ---- Layer-2 GEMM: (32 x 2304) x (2304 x 128); wave -> 2 M x 1 N ----
    {
        const int nt = wave;         // 0..7
        v8f c0 = {}, c1 = {};
        const _Float16* ar0 = A2 + (size_t)nlow * A2S + hi * 8;
        const _Float16* ar1 = A2 + (size_t)(16 + nlow) * A2S + hi * 8;
#pragma unroll 2
        for (int ko = 0; ko < KO2; ++ko) {
            v16h a0 = load_frag16(ar0 + ko * 32);
            v16h a1 = load_frag16(ar1 + ko * 32);
            v16h bm = W2f[(ko * 8 + nt) * 32 + lane];
            c0 = wmma16(a0, bm, c0);
            c1 = wmma16(a1, bm, c1);
        }
        // Residual add + store: lane owns column c and 8 consecutive t.
        const int c = nt * 16 + nlow;
        const size_t base0 = ((size_t)b * CC + c) * TT + t0 + hi * 8;
        const size_t base1 = base0 + 16;      // second M-tile
        {
            const float4* up = (const float4*)(U + base0);
            float4* op = (float4*)(out + base0);
            float4 u0 = up[0], u1 = up[1];
            op[0] = make_float4(u0.x + c0[0], u0.y + c0[1], u0.z + c0[2], u0.w + c0[3]);
            op[1] = make_float4(u1.x + c0[4], u1.y + c0[5], u1.z + c0[6], u1.w + c0[7]);
        }
        {
            const float4* up = (const float4*)(U + base1);
            float4* op = (float4*)(out + base1);
            float4 u0 = up[0], u1 = up[1];
            op[0] = make_float4(u0.x + c1[0], u0.y + c1[1], u0.z + c1[2], u0.w + c1[3]);
            op[1] = make_float4(u1.x + c1[4], u1.y + c1[5], u1.z + c1[6], u1.w + c1[7]);
        }
    }
}

// ---------------------------------------------------------------------------
extern "C" void kernel_launch(void* const* d_in, const int* in_sizes, int n_in,
                              void* d_out, int out_size, void* d_ws, size_t ws_size,
                              hipStream_t stream) {
    (void)in_sizes; (void)n_in; (void)out_size; (void)ws_size;
    const float* U    = (const float*)d_in[0];
    const float* ln_w = (const float*)d_in[1];
    const float* ln_b = (const float*)d_in[2];
    const float* bw1  = (const float*)d_in[3];
    const float* sw1  = (const float*)d_in[4];
    const float* ss1  = (const float*)d_in[5];
    const float* bw2  = (const float*)d_in[6];
    const float* sw2  = (const float*)d_in[7];
    const float* ss2  = (const float*)d_in[8];
    float* out = (float*)d_out;

    // Workspace: mu[64] | rsig[64] | pad | W1f (f16) | W2f (f16)
    float* mu   = (float*)d_ws;
    float* rsig = mu + 64;
    _Float16* W1f = (_Float16*)((char*)d_ws + 1024);
    _Float16* W2f = W1f + (size_t)KO1 * 16 * 32 * 16;   // 294912 halfs each

    stats_kernel<<<BB, 256, 0, stream>>>(U, mu, rsig);
    prep_w1<<<(KO1 * 16 * 32 + 255) / 256, 256, 0, stream>>>(bw1, sw1, ss1, W1f);
    prep_w2<<<(KO2 * 8 * 32 + 255) / 256, 256, 0, stream>>>(bw2, sw2, ss2, W2f);

    hipFuncSetAttribute((const void*)kan_main,
                        hipFuncAttributeMaxDynamicSharedMemorySize,
                        (int)LDS_TOTAL);
    const int nblocks = (BB * TT) / MT;   // 1024 row tiles
    kan_main<<<nblocks, 256, (size_t)LDS_TOTAL, stream>>>(
        U, ln_w, ln_b, mu, rsig, (const v16h*)W1f, (const v16h*)W2f, out);
}
